// MultipleKmeans_90623809946384
// MI455X (gfx1250) — matrix operand order repaired and checked by reading, not compiled, not run on hardware
//
#include <hip/hip_runtime.h>
#include <hip/hip_bf16.h>

// Problem constants (WavLM-Large multi-kmeans quantizer)
#define T_FRAMES 16384
#define E_DIM    1024
#define M_MODELS 8
#define K_CLUST  512

typedef __attribute__((ext_vector_type(16))) __bf16 v16bf;
typedef __attribute__((ext_vector_type(8)))  float  v8f;

__device__ __forceinline__ unsigned int pack_bf16(float a, float b) {
    unsigned int ua = __float_as_uint(a);
    unsigned int ub = __float_as_uint(b);
    // round-to-nearest-even truncation to bf16
    ua = (ua + 0x7FFFu + ((ua >> 16) & 1u)) >> 16;
    ub = (ub + 0x7FFFu + ((ub >> 16) & 1u)) >> 16;
    return (ub << 16) | (ua & 0xFFFFu);   // low 16 = even index (K even), high = odd
}

// ---------------------------------------------------------------------------
// Kernel 0: zero the per-model counters
// ---------------------------------------------------------------------------
__global__ void k_zero_counts(int* counts) {
    if (threadIdx.x < M_MODELS) counts[threadIdx.x] = 0;
}

// ---------------------------------------------------------------------------
// Kernel 1: centroids f32 -> bf16 (packed) + c_sq = ||c||^2 per centroid
// grid = M*K blocks, 128 threads
// ---------------------------------------------------------------------------
__global__ void k_prep_centroids(const float* __restrict__ cent,
                                 unsigned int* __restrict__ centb,
                                 float* __restrict__ c_sq) {
    const int c    = blockIdx.x;                 // global centroid id [0, M*K)
    const int tid  = threadIdx.x;
    const int lane = tid & 31;
    const int wave = tid >> 5;

    const float*  src = cent  + ((size_t)c << 10);   // E_DIM floats
    unsigned int* dst = centb + ((size_t)c << 9);    // E_DIM/2 packed uints

    float ss = 0.0f;
    #pragma unroll 2
    for (int p = tid; p < (E_DIM / 2); p += 128) {
        float f0 = src[2 * p];
        float f1 = src[2 * p + 1];
        ss = fmaf(f0, f0, fmaf(f1, f1, ss));
        dst[p] = pack_bf16(f0, f1);
    }
    // reduce ss over the wave, then across the 4 waves
    #pragma unroll
    for (int m = 16; m >= 1; m >>= 1) ss += __shfl_xor(ss, m, 32);

    __shared__ float part[4];
    if (lane == 0) part[wave] = ss;
    __syncthreads();
    if (tid == 0) c_sq[c] = part[0] + part[1] + part[2] + part[3];
}

// ---------------------------------------------------------------------------
// Kernel 2: bin frames by their chosen model (atomic scatter)
// ---------------------------------------------------------------------------
__global__ void k_bin_frames(const int* __restrict__ model_idx,
                             int* __restrict__ counts,
                             int* __restrict__ buckets) {
    const int t = blockIdx.x * blockDim.x + threadIdx.x;
    if (t < T_FRAMES) {
        const int m = model_idx[t];
        const int p = atomicAdd(&counts[m], 1);
        buckets[m * T_FRAMES + p] = t;
    }
}

// ---------------------------------------------------------------------------
// Kernel 3: main — per (model, 16-frame tile): bf16 WMMA dots vs 512 centroids,
// argmin(c_sq - 2*dot), gather winning centroid row to output.
// K-outer / N-inner: each A fragment is loaded from LDS once and reused for
// 8 WMMAs against 8 live accumulators (one per 16-centroid N-tile).
// block = 128 threads (4 wave32s); grid = (M, T/16)
// ---------------------------------------------------------------------------
__global__ void __launch_bounds__(128)
k_kmeans_main(const float* __restrict__ emb,
              const float* __restrict__ cent,         // f32 centroids [M*K][E]
              const unsigned int* __restrict__ centb, // bf16 packed   [M*K][E/2]
              const float* __restrict__ c_sq,         // [M*K]
              const int* __restrict__ counts,
              const int* __restrict__ buckets,
              float* __restrict__ outp) {
    const int m    = blockIdx.x;
    const int tile = blockIdx.y;
    const int cnt  = counts[m];
    const int base = tile * 16;
    if (base >= cnt) return;

    const int tid  = threadIdx.x;
    const int lane = tid & 31;
    const int wave = tid >> 5;

    __shared__ unsigned int Abuf[16 * (E_DIM / 2)];  // 16 rows x 1024 bf16 = 32 KB
    __shared__ int   sFrames[16];
    __shared__ float sBestV[4][16];
    __shared__ int   sBestI[4][16];
    __shared__ int   sCodes[16];

    if (tid < 16) {
        const int pos = base + tid;
        sFrames[tid] = (pos < cnt) ? buckets[m * T_FRAMES + pos] : -1;
    }
    __syncthreads();

    // ---- gather 16 emb rows into LDS as packed bf16 ------------------------
    for (int idx = tid; idx < 16 * (E_DIM / 2); idx += 128) {
        const int row = idx >> 9;          // / 512
        const int p   = idx & 511;         // pair index within row
        int fr = sFrames[row];
        if (fr < 0) fr = sFrames[0];       // pad rows: recompute a valid frame
        const float2 f2 = *(const float2*)(emb + ((size_t)fr << 10) + 2 * p);
        Abuf[(row << 9) + p] = pack_bf16(f2.x, f2.y);
    }
    __syncthreads();

    // ---- WMMA: this wave owns centroid chunk [wave*128, wave*128+128) ------
    const int colB   = lane & 15;          // N column within 16-wide tile
    const int halfHi = lane >> 4;          // 0: lanes 0-15, 1: lanes 16-31
    const int aRow   = colB;               // A row = lane & 15 (per ISA layout)
    const int nWave  = wave * 128;         // wave's centroid chunk base

    // per-lane base into packed bf16 centroid rows; nt advances columns by 16
    // -> byte stride 16*E_DIM*2 = 32 KB, folds into the load's immediate offset
    const unsigned int* Bbase =
        centb + (((size_t)m * K_CLUST) << 9)
              + (((size_t)(nWave + colB)) << 9) + (halfHi << 2);
    const unsigned int* AbaseRow = &Abuf[aRow << 9];

    // warm L2/near cache for this wave's 8 column groups
    #pragma unroll
    for (int nt = 0; nt < 8; ++nt)
        __builtin_prefetch(Bbase + ((size_t)nt << 13), 0, 1);  // global_prefetch_b8

    v8f acc[8];
    #pragma unroll
    for (int nt = 0; nt < 8; ++nt) acc[nt] = (v8f){};

    #pragma unroll 2
    for (int ks = 0; ks < 32; ++ks) {
        // A fragment (shared by all 8 N-tiles this iteration):
        // row = lane&15; K chunks [ks*32 + h*8, +8) and +16
        union { uint4 u[2]; v16bf v; } af;
        const unsigned int* ap = AbaseRow + (ks << 4) + (halfHi << 2);
        af.u[0] = *(const uint4*)(ap);
        af.u[1] = *(const uint4*)(ap + 8);

        #pragma unroll
        for (int nt = 0; nt < 8; ++nt) {
            // B fragment: lane = column, K chunks [ks*32 + h*8, +8) and +16
            union { uint4 u[2]; v16bf v; } bf;
            const unsigned int* bp = Bbase + ((size_t)nt << 13) + (ks << 4);
            bf.u[0] = *(const uint4*)(bp);
            bf.u[1] = *(const uint4*)(bp + 8);
            acc[nt] = __builtin_amdgcn_wmma_f32_16x16x32_bf16(
                false, af.v, false, bf.v, (short)0, acc[nt], false, false);
        }
    }

    // ---- score = c_sq - 2*dot ; C/D layout: VGPR r -> row r+8*halfHi, col=lane&15
    float bestv[8];
    int   besti[8];
    #pragma unroll
    for (int r = 0; r < 8; ++r) { bestv[r] = 3.4e38f; besti[r] = 0; }

    #pragma unroll
    for (int nt = 0; nt < 8; ++nt) {
        const int nidx  = nWave + nt * 16 + colB;
        const float csq = c_sq[m * K_CLUST + nidx];
        #pragma unroll
        for (int r = 0; r < 8; ++r) {
            const float score = fmaf(-2.0f, acc[nt][r], csq);
            if (score < bestv[r] || (score == bestv[r] && nidx < besti[r])) {
                bestv[r] = score; besti[r] = nidx;
            }
        }
    }

    // ---- per-row min across the 16 columns held by this half-wave ----------
    #pragma unroll
    for (int r = 0; r < 8; ++r) {
        float bv = bestv[r];
        int   bi = besti[r];
        #pragma unroll
        for (int msk = 1; msk < 16; msk <<= 1) {
            const float ov = __shfl_xor(bv, msk, 32);
            const int   oi = __shfl_xor(bi, msk, 32);
            if (ov < bv || (ov == bv && oi < bi)) { bv = ov; bi = oi; }
        }
        if (colB == 0) {
            const int row = r + (halfHi << 3);
            sBestV[wave][row] = bv;
            sBestI[wave][row] = bi;
        }
    }
    __syncthreads();

    // ---- combine the 4 waves' 128-centroid chunks --------------------------
    if (tid < 16) {
        float bv = sBestV[0][tid];
        int   bi = sBestI[0][tid];
        #pragma unroll
        for (int w = 1; w < 4; ++w) {
            const float ov = sBestV[w][tid];
            const int   oi = sBestI[w][tid];
            if (ov < bv || (ov == bv && oi < bi)) { bv = ov; bi = oi; }
        }
        sCodes[tid] = bi;
    }
    __syncthreads();

    // ---- gather winning f32 centroid rows to output (float4, coalesced) ----
    for (int idx = tid; idx < 16 * (E_DIM / 4); idx += 128) {
        const int row = idx >> 8;          // / 256
        const int e4  = idx & 255;
        const int fr  = sFrames[row];
        if (fr >= 0) {
            const float4 v = *(const float4*)(
                cent + (((size_t)(m * K_CLUST + sCodes[row])) << 10) + (e4 << 2));
            *(float4*)(outp + (((size_t)fr) << 10) + (e4 << 2)) = v;
        }
    }
}

// ---------------------------------------------------------------------------
extern "C" void kernel_launch(void* const* d_in, const int* in_sizes, int n_in,
                              void* d_out, int out_size, void* d_ws, size_t ws_size,
                              hipStream_t stream) {
    const float* emb  = (const float*)d_in[0];   // [1,T,E] f32
    const float* cent = (const float*)d_in[1];   // [M,K,E] f32
    const int*   midx = (const int*)d_in[2];     // [T] i32
    float*       outp = (float*)d_out;           // [1,T,E] f32

    char* ws = (char*)d_ws;
    // workspace layout
    unsigned int* centb = (unsigned int*)ws;                       // M*K*E bf16 = 8 MiB
    size_t off = (size_t)M_MODELS * K_CLUST * E_DIM * 2;
    float* c_sq = (float*)(ws + off);                              // M*K f32 = 16 KiB
    off += (size_t)M_MODELS * K_CLUST * 4;
    int* counts = (int*)(ws + off);                                // M ints (padded)
    off += 256;
    int* buckets = (int*)(ws + off);                               // M*T ints = 512 KiB

    k_zero_counts<<<1, 32, 0, stream>>>(counts);
    k_prep_centroids<<<M_MODELS * K_CLUST, 128, 0, stream>>>(cent, centb, c_sq);
    k_bin_frames<<<T_FRAMES / 256, 256, 0, stream>>>(midx, counts, buckets);
    k_kmeans_main<<<dim3(M_MODELS, T_FRAMES / 16), 128, 0, stream>>>(
        emb, cent, centb, c_sq, counts, buckets, outp);
    (void)in_sizes; (void)n_in; (void)out_size; (void)ws_size;
}
